// SparseMoeBlock_63891933495370
// MI455X (gfx1250) — matrix operand order repaired and checked by reading, not compiled, not run on hardware
//
#include <hip/hip_runtime.h>
#include <cstdint>

#define NUM_EXPERTS 32
#define TOPK 4
#define H_DIM 2048
#define I_DIM 1024
#define TILE_M 32
#define THREADS 512           // 16 waves per workgroup

typedef __bf16 bf16;
typedef __attribute__((ext_vector_type(16))) __bf16 v16bf;
typedef __attribute__((ext_vector_type(8)))  __bf16 v8bf;
typedef __attribute__((ext_vector_type(4)))  __bf16 v4bf;
typedef __attribute__((ext_vector_type(16))) float  v16f;
typedef __attribute__((ext_vector_type(4)))  float  v4f;
typedef __attribute__((ext_vector_type(8)))  float  v8f;

// ---------------------------------------------------------------- helpers
__device__ __forceinline__ v8f wmma_bf16(v16bf a, v16bf b, v8f c) {
  return __builtin_amdgcn_wmma_f32_16x16x32_bf16(
      /*neg_a=*/false, a, /*neg_b=*/false, b,
      /*c_mod=*/(short)0, c, /*reuse_a=*/false, /*reuse_b=*/false);
}

// B fragment: lane holds 16 contiguous K values of one W row.
// Native vector conversion -> v_cvt_pk_bf16_f32 pairs.
__device__ __forceinline__ v16bf load_b_frag(const float* __restrict__ p) {
  const v16f f = *(const v16f*)p;                 // 4x global_load_b128
  return __builtin_convertvector(f, v16bf);
}

// A fragment from LDS (row-major bf16): K segment [kb+seg, +16) contiguous
__device__ __forceinline__ v16bf load_a_frag(const bf16* s, int ld, int row, int kb, int seg) {
  const bf16* p = s + row * ld + kb + seg;
  v8bf lo = *(const v8bf*)p;                      // ds_load_b128
  v8bf hi = *(const v8bf*)(p + 8);                // ds_load_b128
  return __builtin_shufflevector(lo, hi, 0,1,2,3,4,5,6,7,8,9,10,11,12,13,14,15);
}

// ---------------------------------------------------------------- kernels
__global__ void zero_kernel(float* __restrict__ out, long long n, int* __restrict__ counts) {
  long long i = (long long)blockIdx.x * blockDim.x + threadIdx.x;
  if (i < NUM_EXPERTS) counts[i] = 0;
  long long stride = (long long)gridDim.x * blockDim.x;
  for (; i < n; i += stride) out[i] = 0.f;
}

// One wave32 per token; lane == expert. Softmax + top-4 + normalize.
__global__ void router_kernel(const float* __restrict__ x, const float* __restrict__ rw,
                              int* __restrict__ topk_idx, float* __restrict__ topk_w,
                              int* __restrict__ counts, int T) {
  const int wave = threadIdx.x >> 5;
  const int lane = threadIdx.x & 31;
  const int t = blockIdx.x * (blockDim.x >> 5) + wave;
  if (t >= T) return;
  const float* xr = x + (size_t)t * H_DIM;
  const float* wr = rw + (size_t)lane * H_DIM;
  float acc = 0.f;
  for (int h = 0; h < H_DIM; h += 4) {
    v4f xv = *(const v4f*)(xr + h);
    v4f wv = *(const v4f*)(wr + h);
    acc += xv.x * wv.x + xv.y * wv.y + xv.z * wv.z + xv.w * wv.w;
  }
  float m = acc;
  for (int off = 16; off; off >>= 1) m = fmaxf(m, __shfl_xor(m, off, 32));
  float p = __expf(acc - m);
  float ssum = p;
  for (int off = 16; off; off >>= 1) ssum += __shfl_xor(ssum, off, 32);
  float v = p / ssum;     // softmax prob for this lane's expert
  float wsum = 0.f; int sel_i[TOPK]; float sel_w[TOPK];
  #pragma unroll
  for (int k = 0; k < TOPK; ++k) {
    float bv = v; int bi = lane;
    for (int off = 16; off; off >>= 1) {
      float ov = __shfl_xor(bv, off, 32);
      int   oi = __shfl_xor(bi, off, 32);
      if (ov > bv || (ov == bv && oi < bi)) { bv = ov; bi = oi; }
    }
    sel_i[k] = bi; sel_w[k] = bv; wsum += bv;
    if (lane == bi) v = -1.0f;
  }
  const float inv = 1.f / wsum;
  #pragma unroll
  for (int k = 0; k < TOPK; ++k) {
    if (lane == k) {
      topk_idx[t * TOPK + k] = sel_i[k];
      topk_w[t * TOPK + k]   = sel_w[k] * inv;
      atomicAdd(&counts[sel_i[k]], 1);
    }
  }
}

__global__ void scan_kernel(const int* __restrict__ counts, int* __restrict__ offs,
                            int* __restrict__ cursor) {
  if (threadIdx.x == 0) {
    int s = 0;
    for (int e = 0; e < NUM_EXPERTS; ++e) { offs[e] = s; cursor[e] = s; s += counts[e]; }
  }
}

__global__ void scatter_kernel(const int* __restrict__ topk_idx, const float* __restrict__ topk_w,
                               int* __restrict__ cursor, int* __restrict__ sorted_tok,
                               float* __restrict__ sorted_w, int total) {
  int i = blockIdx.x * blockDim.x + threadIdx.x;
  if (i >= total) return;
  int e = topk_idx[i];
  int pos = atomicAdd(&cursor[e], 1);
  sorted_tok[pos] = i >> 2;          // token index
  sorted_w[pos]   = topk_w[i];
}

// Fused gate/up/silu/down grouped-GEMM for one (expert, 32-token tile).
// 16 waves: GEMM1/2 -> each wave owns 64 I-columns; GEMM3 -> 128 H-columns.
__global__ __launch_bounds__(THREADS, 1)
void expert_kernel(const float* __restrict__ x,
                   const float* __restrict__ w_gate,
                   const float* __restrict__ w_up,
                   const float* __restrict__ w_down,
                   const int* __restrict__ sorted_tok,
                   const float* __restrict__ sorted_w,
                   const int* __restrict__ offs,
                   const int* __restrict__ counts,
                   float* __restrict__ out) {
  __shared__ bf16  sA[TILE_M * H_DIM];   // 128 KB gathered activations (bf16)
  __shared__ bf16  sH[TILE_M * I_DIM];   //  64 KB silu(g)*u intermediate (bf16)
  __shared__ int   s_tok[TILE_M];
  __shared__ float s_w[TILE_M];

  const int e   = blockIdx.y;
  const int cnt = counts[e];
  const int t0  = blockIdx.x * TILE_M;
  if (t0 >= cnt) return;
  const int base = offs[e];
  const int rows = (cnt - t0 < TILE_M) ? (cnt - t0) : TILE_M;

  const int tid    = threadIdx.x;
  const int lane   = tid & 31;
  const int wave   = tid >> 5;               // 0..15
  const int lane15 = lane & 15;
  const int segA   = (lane & 16) ? 8 : 0;    // A-matrix K interleave
  const int segB   = (lane & 16) ? 16 : 0;   // B-matrix K split
  const int mrow   = (lane >> 4) << 3;       // +8 row for upper-half lanes (C/D layout)

  if (tid < TILE_M) {
    const bool valid = tid < rows;
    s_tok[tid] = valid ? sorted_tok[base + t0 + tid] : 0;
    s_w[tid]   = valid ? sorted_w[base + t0 + tid]   : 0.f;
  }
  __syncthreads();

  // Stage gathered A tile (fp32 -> bf16) into LDS.
  for (int idx = tid; idx < TILE_M * (H_DIM / 4); idx += THREADS) {
    const int row = idx / (H_DIM / 4);
    const int c4  = idx % (H_DIM / 4);
    const v4f v = *(const v4f*)(x + (size_t)s_tok[row] * H_DIM + c4 * 4);
    *(v4bf*)&sA[row * H_DIM + c4 * 4] = __builtin_convertvector(v, v4bf);
  }
  __syncthreads();

  const float* wg_e = w_gate + (size_t)e * I_DIM * H_DIM;
  const float* wu_e = w_up   + (size_t)e * I_DIM * H_DIM;

  // ---- GEMM 1+2: G = A*Wg^T, U = A*Wu^T ----
  v8f accG[2][4], accU[2][4];
  #pragma unroll
  for (int m = 0; m < 2; ++m)
    #pragma unroll
    for (int it = 0; it < 4; ++it) { accG[m][it] = (v8f)0; accU[m][it] = (v8f)0; }

  for (int kb = 0; kb < H_DIM; kb += 32) {
    const v16bf a0 = load_a_frag(sA, H_DIM, lane15,      kb, segB);
    const v16bf a1 = load_a_frag(sA, H_DIM, 16 + lane15, kb, segB);
    #pragma unroll
    for (int it = 0; it < 4; ++it) {
      const int n = wave * 64 + it * 16 + lane15;
      const float* pg = wg_e + (size_t)n * H_DIM + kb + segB;
      const float* pu = wu_e + (size_t)n * H_DIM + kb + segB;
      __builtin_prefetch(pg + 32, 0, 0);     // global_prefetch_b8: next K chunk
      __builtin_prefetch(pu + 32, 0, 0);
      const v16bf bg = load_b_frag(pg);
      const v16bf bu = load_b_frag(pu);
      accG[0][it] = wmma_bf16(a0, bg, accG[0][it]);
      accG[1][it] = wmma_bf16(a1, bg, accG[1][it]);
      accU[0][it] = wmma_bf16(a0, bu, accU[0][it]);
      accU[1][it] = wmma_bf16(a1, bu, accU[1][it]);
    }
  }

  // silu(g) * u -> sH (bf16)
  #pragma unroll
  for (int m = 0; m < 2; ++m)
    #pragma unroll
    for (int it = 0; it < 4; ++it)
      #pragma unroll
      for (int r = 0; r < 8; ++r) {
        const float g = accG[m][it][r];
        const float u = accU[m][it][r];
        const float hv = (g / (1.f + __expf(-g))) * u;
        const int row = m * 16 + mrow + r;
        const int col = wave * 64 + it * 16 + lane15;
        sH[row * I_DIM + col] = (bf16)hv;
      }
  __syncthreads();

  // ---- GEMM 3: O = H * Wd^T ----
  const float* wd_e = w_down + (size_t)e * H_DIM * I_DIM;
  v8f accO[2][8];
  #pragma unroll
  for (int m = 0; m < 2; ++m)
    #pragma unroll
    for (int ct = 0; ct < 8; ++ct) accO[m][ct] = (v8f)0;

  for (int kb = 0; kb < I_DIM; kb += 32) {
    const v16bf a0 = load_a_frag(sH, I_DIM, lane15,      kb, segB);
    const v16bf a1 = load_a_frag(sH, I_DIM, 16 + lane15, kb, segB);
    #pragma unroll
    for (int ct = 0; ct < 8; ++ct) {
      const int n = wave * 128 + ct * 16 + lane15;
      const float* pd = wd_e + (size_t)n * I_DIM + kb + segB;
      __builtin_prefetch(pd + 32, 0, 0);
      const v16bf bd = load_b_frag(pd);
      accO[0][ct] = wmma_bf16(a0, bd, accO[0][ct]);
      accO[1][ct] = wmma_bf16(a1, bd, accO[1][ct]);
    }
  }

  // Weighted scatter-add into output.
  #pragma unroll
  for (int m = 0; m < 2; ++m)
    #pragma unroll
    for (int ct = 0; ct < 8; ++ct) {
      const int col = wave * 128 + ct * 16 + lane15;
      #pragma unroll
      for (int r = 0; r < 8; ++r) {
        const int row = m * 16 + mrow + r;
        if (row < rows) {
          atomicAdd(out + (size_t)s_tok[row] * H_DIM + col, accO[m][ct][r] * s_w[row]);
        }
      }
    }
}

// ---------------------------------------------------------------- launch
extern "C" void kernel_launch(void* const* d_in, const int* in_sizes, int n_in,
                              void* d_out, int out_size, void* d_ws, size_t ws_size,
                              hipStream_t stream) {
  (void)n_in; (void)out_size; (void)ws_size;
  const float* x  = (const float*)d_in[0];
  const float* rw = (const float*)d_in[1];
  const float* wg = (const float*)d_in[2];
  const float* wu = (const float*)d_in[3];
  const float* wd = (const float*)d_in[4];
  float* out = (float*)d_out;

  const int T = in_sizes[0] / H_DIM;        // 8192 tokens
  const int total = T * TOPK;

  char* w = (char*)d_ws;
  int*   counts     = (int*)(w + 0);
  int*   cursor     = (int*)(w + 128);
  int*   offs       = (int*)(w + 256);
  int*   topk_idx   = (int*)(w + 512);
  float* topk_w     = (float*)(w + 512 + (size_t)total * 4);
  int*   sorted_tok = (int*)(w + 512 + (size_t)total * 8);
  float* sorted_w   = (float*)(w + 512 + (size_t)total * 12);

  zero_kernel<<<4096, 256, 0, stream>>>(out, (long long)T * H_DIM, counts);
  router_kernel<<<(T + 7) / 8, 256, 0, stream>>>(x, rw, topk_idx, topk_w, counts, T);
  scan_kernel<<<1, 32, 0, stream>>>(counts, offs, cursor);
  scatter_kernel<<<(total + 255) / 256, 256, 0, stream>>>(topk_idx, topk_w, cursor,
                                                          sorted_tok, sorted_w, total);
  dim3 grid((T + TILE_M - 1) / TILE_M, NUM_EXPERTS);
  expert_kernel<<<grid, THREADS, 0, stream>>>(x, wg, wu, wd, sorted_tok, sorted_w,
                                              offs, counts, out);
}
